// ImageCaptionModel_53154515255972
// MI455X (gfx1250) — matrix operand order repaired and checked by reading, not compiled, not run on hardware
//
#include <hip/hip_runtime.h>
#include <hip/hip_bf16.h>

// ---------------------------------------------------------------------------
// ImageCaptionModel on MI455X (gfx1250): bf16 WMMA, register-blocked.
//   feat = leaky_relu(features @ Wp + bp)
//   25 steps of 2-layer LSTM; one wave = 4 M-tiles x 16 units x 4 gates fused
//   logits = hs @ Wout + bout : 4x5 tile blocking, non-temporal fp32 stores
//   weight transposes: LDS-tiled (coalesced both sides)
// ---------------------------------------------------------------------------

typedef __attribute__((ext_vector_type(16))) __bf16 v16bf;
typedef __attribute__((ext_vector_type(8)))  float  v8f;

#define DEV __device__ __forceinline__

static constexpr int Bn = 128;   // batch
static constexpr int Tn = 25;    // timesteps
static constexpr int En = 512;   // embedding dim
static constexpr int Hn = 1024;  // hidden dim
static constexpr int Vn = 10000; // vocab
static constexpr int Fn = 2048;  // feature dim

DEV unsigned short f2bf(float x) {              // fp32 -> bf16 (RNE)
  union { float f; unsigned u; } v; v.f = x;
  unsigned r = v.u + 0x7FFFu + ((v.u >> 16) & 1u);
  return (unsigned short)(r >> 16);
}

DEV float sigmoidf(float x) { return 1.0f / (1.0f + __expf(-x)); }

DEV v8f wmma_bf16(v16bf a, v16bf b, v8f c) {
  // emits v_wmma_f32_16x16x32_bf16
  return __builtin_amdgcn_wmma_f32_16x16x32_bf16(false, a, false, b,
                                                 (short)0, c, false, false);
}

// A fragment (16x32 bf16): rowbase = &A[row][0] (row-major, contiguous K).
DEV v16bf load_a_frag(const unsigned short* rowbase, int kbase, int half) {
  union { v16bf v; unsigned u[8]; } f;
  const unsigned* p = (const unsigned*)rowbase;
  int q = (kbase >> 1) + half * 8;
#pragma unroll
  for (int j = 0; j < 4; ++j) f.u[j] = p[q + j];
#pragma unroll
  for (int j = 0; j < 4; ++j) f.u[4 + j] = p[q + 4 + j];
  return f.v;
}

// B fragment (32x16 bf16): weights stored transposed [N][K] bf16.
DEV v16bf load_b_frag(const unsigned short* rowbase, int kbase, int half) {
  union { v16bf v; unsigned u[8]; } f;
  const unsigned* p = (const unsigned*)rowbase;
  int q = (kbase >> 1) + half * 8;
#pragma unroll
  for (int j = 0; j < 8; ++j) f.u[j] = p[q + j];
  return f.v;
}

// ---------------------------------------------------------------- utilities
__global__ void k_cvt_bf16(const float* __restrict__ src,
                           unsigned short* __restrict__ dst, int n) {
  int i = blockIdx.x * blockDim.x + threadIdx.x;
  if (i < n) dst[i] = f2bf(src[i]);
}

// src [K][N] fp32 -> dst [N][K] bf16, LDS-tiled transpose (coalesced R & W).
// blockDim = (32, 8); grid = ((N+31)/32, (K+31)/32)
__global__ void k_trcvt_bf16(const float* __restrict__ src,
                             unsigned short* __restrict__ dst, int K, int N) {
  __shared__ unsigned short tile[32][33];
  int nb = blockIdx.x * 32, kb = blockIdx.y * 32;
  int tx = threadIdx.x, ty = threadIdx.y;
#pragma unroll
  for (int dy = 0; dy < 32; dy += 8) {
    int k = kb + ty + dy, n = nb + tx;
    if (k < K && n < N) tile[ty + dy][tx] = f2bf(src[(size_t)k * N + n]);
  }
  __syncthreads();
#pragma unroll
  for (int dy = 0; dy < 32; dy += 8) {
    int n = nb + ty + dy, k = kb + tx;
    if (n < N && k < K) dst[(size_t)n * K + k] = tile[tx][ty + dy];
  }
}

__global__ void k_zero(unsigned int* __restrict__ p, int n) {
  int i = blockIdx.x * blockDim.x + threadIdx.x;
  if (i < n) p[i] = 0u;
}

// ------------------------------------------------ feature projection (WMMA)
// feat = leaky_relu(features @ Wp + bp), output bf16 [128][1024]; small GEMM.
__global__ void k_featproj(const unsigned short* __restrict__ A,   // [128][2048]
                           const unsigned short* __restrict__ Bt,  // [1024][2048]
                           const float* __restrict__ bias,
                           unsigned short* __restrict__ out) {     // [128][1024]
  const int K = Fn;
  int wave = (blockIdx.x * blockDim.x + threadIdx.x) >> 5;
  int lane = threadIdx.x & 31;
  int mTile = wave & 7, nTile = wave >> 3;       // 8 x 64 tiles, exact
  int r = lane & 15, half = lane >> 4;
  const unsigned short* arow = A + (size_t)(mTile * 16 + r) * K;
  const unsigned short* brow = Bt + (size_t)(nTile * 16 + r) * K;
  v8f acc = {0.f, 0.f, 0.f, 0.f, 0.f, 0.f, 0.f, 0.f};
  for (int k = 0; k < K; k += 32) {
    acc = wmma_bf16(load_a_frag(arow, k, half), load_b_frag(brow, k, half), acc);
  }
  int n = nTile * 16 + r;
  float bv = bias[n];
#pragma unroll
  for (int j = 0; j < 8; ++j) {
    int m = mTile * 16 + half * 8 + j;
    float y = acc[j] + bv;
    y = (y > 0.0f) ? y : 0.01f * y;              // jax leaky_relu slope 0.01
    out[m * Hn + n] = f2bf(y);
  }
}

// ------------------------------------------------ LSTM layer 0 step (WMMA)
// comb = [emb[token], feat, h_prev]  (K = 2560).
// One wave = 4 M-tiles (64 batch rows) x 16 units x 4 gates = 16 accumulators.
// 128 waves total (2 m-groups x 64 n-tiles); 32 blocks x 128 threads.
__global__ void k_lstm0(const int* __restrict__ tokens, int t,
                        const unsigned short* __restrict__ emb,    // [V][512]
                        const unsigned short* __restrict__ feat,   // [B][H]
                        const unsigned short* __restrict__ h_prev, // [B][H]
                        const float* __restrict__ c_prev,          // [B][H]
                        const unsigned short* __restrict__ Wt,     // [4][H][2560]
                        const float* __restrict__ bf_, const float* __restrict__ bi_,
                        const float* __restrict__ bo_, const float* __restrict__ bg_,
                        unsigned short* __restrict__ h_next,
                        float* __restrict__ c_next) {
  const int K = En + Hn + Hn;                    // 2560
  int wave = (blockIdx.x * blockDim.x + threadIdx.x) >> 5;
  int lane = threadIdx.x & 31;
  int mg = wave & 1, nt = wave >> 1;             // 2 m-groups x 64 n-tiles
  int r = lane & 15, half = lane >> 4;

  const unsigned short* aX[4];
  const unsigned short* aF[4];
  const unsigned short* aH[4];
#pragma unroll
  for (int mi = 0; mi < 4; ++mi) {
    int mrow = mg * 64 + mi * 16 + r;
    int tok = tokens[mrow * Tn + t];
    aX[mi] = emb + (size_t)tok * En;
    aF[mi] = feat + (size_t)mrow * Hn;
    aH[mi] = h_prev + (size_t)mrow * Hn;
  }
  int n = nt * 16 + r;
  const unsigned short* bW[4];
#pragma unroll
  for (int g = 0; g < 4; ++g) bW[g] = Wt + (size_t)(g * Hn + n) * K;

  v8f acc[4][4];                                  // [gate][mtile]
#pragma unroll
  for (int g = 0; g < 4; ++g)
#pragma unroll
    for (int mi = 0; mi < 4; ++mi)
      acc[g][mi] = (v8f){0.f,0.f,0.f,0.f,0.f,0.f,0.f,0.f};

  // segment 1: token embedding (weight K 0..511)
  for (int k = 0; k < En; k += 32) {
    __builtin_prefetch((const void*)(bW[0] + k + 512), 0, 0);
    __builtin_prefetch((const void*)(bW[3] + k + 512), 0, 0);
    v16bf a[4];
#pragma unroll
    for (int mi = 0; mi < 4; ++mi) a[mi] = load_a_frag(aX[mi], k, half);
#pragma unroll
    for (int g = 0; g < 4; ++g) {
      v16bf b = load_b_frag(bW[g], k, half);
#pragma unroll
      for (int mi = 0; mi < 4; ++mi) acc[g][mi] = wmma_bf16(a[mi], b, acc[g][mi]);
    }
  }
  // segment 2: projected features (weight K 512..1535)
  for (int k = 0; k < Hn; k += 32) {
    int kw = En + k;
    __builtin_prefetch((const void*)(bW[0] + kw + 512), 0, 0);
    __builtin_prefetch((const void*)(bW[3] + kw + 512), 0, 0);
    v16bf a[4];
#pragma unroll
    for (int mi = 0; mi < 4; ++mi) a[mi] = load_a_frag(aF[mi], k, half);
#pragma unroll
    for (int g = 0; g < 4; ++g) {
      v16bf b = load_b_frag(bW[g], kw, half);
#pragma unroll
      for (int mi = 0; mi < 4; ++mi) acc[g][mi] = wmma_bf16(a[mi], b, acc[g][mi]);
    }
  }
  // segment 3: recurrent h (weight K 1536..2559)
  for (int k = 0; k < Hn; k += 32) {
    int kw = En + Hn + k;
    v16bf a[4];
#pragma unroll
    for (int mi = 0; mi < 4; ++mi) a[mi] = load_a_frag(aH[mi], k, half);
#pragma unroll
    for (int g = 0; g < 4; ++g) {
      v16bf b = load_b_frag(bW[g], kw, half);
#pragma unroll
      for (int mi = 0; mi < 4; ++mi) acc[g][mi] = wmma_bf16(a[mi], b, acc[g][mi]);
    }
  }

  float vbf = bf_[n], vbi = bi_[n], vbo = bo_[n], vbg = bg_[n];
#pragma unroll
  for (int mi = 0; mi < 4; ++mi) {
#pragma unroll
    for (int j = 0; j < 8; ++j) {
      int m = mg * 64 + mi * 16 + half * 8 + j;
      float fg = sigmoidf(acc[0][mi][j] + vbf);
      float ig = sigmoidf(acc[1][mi][j] + vbi);
      float og = sigmoidf(acc[2][mi][j] + vbo);
      float gg = tanhf(acc[3][mi][j] + vbg);
      float c = fg * c_prev[m * Hn + n] + ig * gg;
      c_next[m * Hn + n] = c;
      h_next[m * Hn + n] = f2bf(og * tanhf(c));
    }
  }
}

// ------------------------------------------------ LSTM layer 1 step (WMMA)
// comb = [h0 (this step), h1_prev] (K = 2048); h out goes straight into hs.
__global__ void k_lstm1(const unsigned short* __restrict__ h0,     // [B][H]
                        const unsigned short* __restrict__ h_prev, // [B][H]
                        const float* __restrict__ c_prev,
                        const unsigned short* __restrict__ Wt,     // [4][H][2048]
                        const float* __restrict__ bf_, const float* __restrict__ bi_,
                        const float* __restrict__ bo_, const float* __restrict__ bg_,
                        unsigned short* __restrict__ h_next,       // hs + t*B*H
                        float* __restrict__ c_next) {
  const int K = Hn + Hn;                         // 2048
  int wave = (blockIdx.x * blockDim.x + threadIdx.x) >> 5;
  int lane = threadIdx.x & 31;
  int mg = wave & 1, nt = wave >> 1;
  int r = lane & 15, half = lane >> 4;

  const unsigned short* aA[4];
  const unsigned short* aH[4];
#pragma unroll
  for (int mi = 0; mi < 4; ++mi) {
    int mrow = mg * 64 + mi * 16 + r;
    aA[mi] = h0 + (size_t)mrow * Hn;
    aH[mi] = h_prev + (size_t)mrow * Hn;
  }
  int n = nt * 16 + r;
  const unsigned short* bW[4];
#pragma unroll
  for (int g = 0; g < 4; ++g) bW[g] = Wt + (size_t)(g * Hn + n) * K;

  v8f acc[4][4];
#pragma unroll
  for (int g = 0; g < 4; ++g)
#pragma unroll
    for (int mi = 0; mi < 4; ++mi)
      acc[g][mi] = (v8f){0.f,0.f,0.f,0.f,0.f,0.f,0.f,0.f};

  for (int k = 0; k < Hn; k += 32) {
    __builtin_prefetch((const void*)(bW[0] + k + 512), 0, 0);
    __builtin_prefetch((const void*)(bW[3] + k + 512), 0, 0);
    v16bf a[4];
#pragma unroll
    for (int mi = 0; mi < 4; ++mi) a[mi] = load_a_frag(aA[mi], k, half);
#pragma unroll
    for (int g = 0; g < 4; ++g) {
      v16bf b = load_b_frag(bW[g], k, half);
#pragma unroll
      for (int mi = 0; mi < 4; ++mi) acc[g][mi] = wmma_bf16(a[mi], b, acc[g][mi]);
    }
  }
  for (int k = 0; k < Hn; k += 32) {
    int kw = Hn + k;
    v16bf a[4];
#pragma unroll
    for (int mi = 0; mi < 4; ++mi) a[mi] = load_a_frag(aH[mi], k, half);
#pragma unroll
    for (int g = 0; g < 4; ++g) {
      v16bf b = load_b_frag(bW[g], kw, half);
#pragma unroll
      for (int mi = 0; mi < 4; ++mi) acc[g][mi] = wmma_bf16(a[mi], b, acc[g][mi]);
    }
  }

  float vbf = bf_[n], vbi = bi_[n], vbo = bo_[n], vbg = bg_[n];
#pragma unroll
  for (int mi = 0; mi < 4; ++mi) {
#pragma unroll
    for (int j = 0; j < 8; ++j) {
      int m = mg * 64 + mi * 16 + half * 8 + j;
      float fg = sigmoidf(acc[0][mi][j] + vbf);
      float ig = sigmoidf(acc[1][mi][j] + vbi);
      float og = sigmoidf(acc[2][mi][j] + vbo);
      float gg = tanhf(acc[3][mi][j] + vbg);
      float c = fg * c_prev[m * Hn + n] + ig * gg;
      c_next[m * Hn + n] = c;
      h_next[m * Hn + n] = f2bf(og * tanhf(c));
    }
  }
}

// ------------------------------------------------ output logits GEMM (WMMA)
// hs [T*B][H] (row = t*B+b, bf16) @ WoutT [V][H] -> out [B][T][V] fp32 (NT)
// 4 M-tiles x 5 N-tiles per wave: 50 x 125 = 6250 waves; 3125 blocks x 64.
__global__ void k_logits(const unsigned short* __restrict__ hs,
                         const unsigned short* __restrict__ WoutT,
                         const float* __restrict__ bout,
                         float* __restrict__ out) {
  int wave = (blockIdx.x * blockDim.x + threadIdx.x) >> 5;
  int lane = threadIdx.x & 31;
  int mg = wave % 50, ng = wave / 50;            // 50 x 125, exact
  int r = lane & 15, half = lane >> 4;

  const unsigned short* arow[4];
#pragma unroll
  for (int mi = 0; mi < 4; ++mi)
    arow[mi] = hs + (size_t)(mg * 64 + mi * 16 + r) * Hn;
  const unsigned short* brow[5];
#pragma unroll
  for (int nj = 0; nj < 5; ++nj)
    brow[nj] = WoutT + (size_t)(ng * 80 + nj * 16 + r) * Hn;

  v8f acc[5][4];
#pragma unroll
  for (int nj = 0; nj < 5; ++nj)
#pragma unroll
    for (int mi = 0; mi < 4; ++mi)
      acc[nj][mi] = (v8f){0.f,0.f,0.f,0.f,0.f,0.f,0.f,0.f};

  for (int k = 0; k < Hn; k += 32) {
    v16bf a[4];
#pragma unroll
    for (int mi = 0; mi < 4; ++mi) a[mi] = load_a_frag(arow[mi], k, half);
#pragma unroll
    for (int nj = 0; nj < 5; ++nj) {
      v16bf b = load_b_frag(brow[nj], k, half);
#pragma unroll
      for (int mi = 0; mi < 4; ++mi) acc[nj][mi] = wmma_bf16(a[mi], b, acc[nj][mi]);
    }
  }

#pragma unroll
  for (int nj = 0; nj < 5; ++nj) {
    int n = ng * 80 + nj * 16 + r;
    float bv = bout[n];
#pragma unroll
    for (int mi = 0; mi < 4; ++mi) {
#pragma unroll
      for (int j = 0; j < 8; ++j) {
        int row = mg * 64 + mi * 16 + half * 8 + j;   // row = t*B + b
        int tt = row >> 7;                            // / 128
        int bb = row & 127;
        __builtin_nontemporal_store(acc[nj][mi][j] + bv,
                                    out + ((size_t)bb * Tn + tt) * Vn + n);
      }
    }
  }
}

// ---------------------------------------------------------------------------
extern "C" void kernel_launch(void* const* d_in, const int* in_sizes, int n_in,
                              void* d_out, int out_size, void* d_ws, size_t ws_size,
                              hipStream_t stream) {
  const int*   tokens    = (const int*)  d_in[0];
  const float* features  = (const float*)d_in[1];
  const float* embedding = (const float*)d_in[2];
  const float* Wp        = (const float*)d_in[3];
  const float* bp        = (const float*)d_in[4];
  const float* W0[4] = {(const float*)d_in[5],  (const float*)d_in[7],
                        (const float*)d_in[9],  (const float*)d_in[11]};
  const float* b0[4] = {(const float*)d_in[6],  (const float*)d_in[8],
                        (const float*)d_in[10], (const float*)d_in[12]};
  const float* W1[4] = {(const float*)d_in[13], (const float*)d_in[15],
                        (const float*)d_in[17], (const float*)d_in[19]};
  const float* b1[4] = {(const float*)d_in[14], (const float*)d_in[16],
                        (const float*)d_in[18], (const float*)d_in[20]};
  const float* Wout = (const float*)d_in[21];
  const float* bout = (const float*)d_in[22];
  float* out = (float*)d_out;

  // ---- workspace carve-out (all sizes multiples of 256 bytes) ----
  size_t off = 0;
  auto take = [&](size_t bytes) -> void* {
    void* p = (char*)d_ws + off;
    off += (bytes + 255) & ~(size_t)255;
    return p;
  };
  unsigned short* embB  = (unsigned short*)take((size_t)Vn * En * 2);
  unsigned short* featsB= (unsigned short*)take((size_t)Bn * Fn * 2);
  unsigned short* WpT   = (unsigned short*)take((size_t)Hn * Fn * 2);
  unsigned short* W0T   = (unsigned short*)take((size_t)4 * Hn * 2560 * 2);
  unsigned short* W1T   = (unsigned short*)take((size_t)4 * Hn * 2048 * 2);
  unsigned short* WoutT = (unsigned short*)take((size_t)Vn * Hn * 2);
  unsigned short* featB = (unsigned short*)take((size_t)Bn * Hn * 2);
  // contiguous zero-init region: h0[2] bf16, h1zero bf16, c0[2] f32, c1[2] f32
  unsigned short* h0    = (unsigned short*)take((size_t)2 * Bn * Hn * 2);
  unsigned short* h1z   = (unsigned short*)take((size_t)Bn * Hn * 2);
  float*          c0    = (float*)take((size_t)2 * Bn * Hn * 4);
  float*          c1    = (float*)take((size_t)2 * Bn * Hn * 4);
  unsigned short* hs    = (unsigned short*)take((size_t)Tn * Bn * Hn * 2);
  (void)ws_size; (void)in_sizes; (void)n_in; (void)out_size;

  const int thr = 256;
  dim3 tb(32, 8);
  // ---- preprocessing: bf16 conversion / LDS-tiled weight transposition ----
  k_cvt_bf16<<<(Vn * En + thr - 1) / thr, thr, 0, stream>>>(embedding, embB, Vn * En);
  k_cvt_bf16<<<(Bn * Fn + thr - 1) / thr, thr, 0, stream>>>(features, featsB, Bn * Fn);
  k_trcvt_bf16<<<dim3((Hn + 31) / 32, (Fn + 31) / 32), tb, 0, stream>>>(Wp, WpT, Fn, Hn);
  for (int g = 0; g < 4; ++g) {
    k_trcvt_bf16<<<dim3((Hn + 31) / 32, (2560 + 31) / 32), tb, 0, stream>>>(
        W0[g], W0T + (size_t)g * Hn * 2560, 2560, Hn);
    k_trcvt_bf16<<<dim3((Hn + 31) / 32, (2048 + 31) / 32), tb, 0, stream>>>(
        W1[g], W1T + (size_t)g * Hn * 2048, 2048, Hn);
  }
  k_trcvt_bf16<<<dim3((Vn + 31) / 32, (Hn + 31) / 32), tb, 0, stream>>>(Wout, WoutT, Hn, Vn);

  // zero h0/h1z/c0/c1 (contiguous region)
  const int zn = (2 * Bn * Hn * 2 + Bn * Hn * 2 + 2 * Bn * Hn * 4 + 2 * Bn * Hn * 4) / 4;
  k_zero<<<(zn + thr - 1) / thr, thr, 0, stream>>>((unsigned int*)h0, zn);

  // ---- feature projection ----
  k_featproj<<<128, 128, 0, stream>>>(featsB, WpT, bp, featB);

  // ---- 25 LSTM steps (2 layers, WMMA-fused gates, 4x M-blocking) ----
  for (int t = 0; t < Tn; ++t) {
    int pr = t & 1, nx = 1 - pr;
    unsigned short* h0p = h0 + (size_t)pr * Bn * Hn;
    unsigned short* h0n = h0 + (size_t)nx * Bn * Hn;
    float* c0p = c0 + (size_t)pr * Bn * Hn;
    float* c0n = c0 + (size_t)nx * Bn * Hn;
    k_lstm0<<<32, 128, 0, stream>>>(tokens, t, embB, featB, h0p, c0p, W0T,
                                    b0[0], b0[1], b0[2], b0[3], h0n, c0n);
    const unsigned short* h1p = (t == 0) ? h1z : hs + (size_t)(t - 1) * Bn * Hn;
    float* c1p = c1 + (size_t)pr * Bn * Hn;
    float* c1n = c1 + (size_t)nx * Bn * Hn;
    k_lstm1<<<32, 128, 0, stream>>>(h0n, h1p, c1p, W1T,
                                    b1[0], b1[1], b1[2], b1[3],
                                    hs + (size_t)t * Bn * Hn, c1n);
  }

  // ---- output logits: 6250 waves, 20 WMMA tiles each ----
  k_logits<<<3125, 64, 0, stream>>>(hs, WoutT, bout, out);
}
